// LSTMModel_24979529793785
// MI455X (gfx1250) — matrix operand order repaired and checked by reading, not compiled, not run on hardware
//
#include <hip/hip_runtime.h>
#include <hip/hip_bf16.h>

#define VOCAB 50000
#define EMBED 300
#define EPAD  320          // EMBED padded to multiple of 32 (WMMA K)
#define UNITS 512
#define NGATE 2048         // 4*UNITS, gate order i,f,g,o
#define SEQ   512
#define BATCH 256
#define TB    (SEQ*BATCH)  // 131072 rows of the input-projection GEMM

#define LDSP_G (EPAD + 8)   // padded LDS row stride (bf16 elems), bank-spread
#define LDSP_S (UNITS + 8)

#define USE_ASYNC_LDS 1

typedef __bf16 bf16;
typedef __attribute__((ext_vector_type(16))) __bf16 v16bf;
typedef __attribute__((ext_vector_type(8)))  __bf16 v8bf;
typedef __attribute__((ext_vector_type(8)))  float  v8f;
typedef __attribute__((ext_vector_type(4)))  int    i32x4;

// ---------------------------------------------------------------------------
// 16-byte global->LDS staging chunk via the CDNA5 async-to-LDS path (ASYNCcnt).
// ---------------------------------------------------------------------------
__device__ __forceinline__ void cp_b128(bf16* sdst, const bf16* gsrc) {
#if USE_ASYNC_LDS && __has_builtin(__builtin_amdgcn_global_load_async_to_lds_b128)
  __builtin_amdgcn_global_load_async_to_lds_b128(
      (i32x4*)gsrc, (i32x4*)sdst, 0, 0);
#else
  *(v8bf*)sdst = *(const v8bf*)gsrc;
#endif
}

__device__ __forceinline__ void cp_wait() {
#if USE_ASYNC_LDS && __has_builtin(__builtin_amdgcn_global_load_async_to_lds_b128)
#if __has_builtin(__builtin_amdgcn_s_wait_asynccnt)
  __builtin_amdgcn_s_wait_asynccnt(0);
#else
  asm volatile("s_wait_asynccnt 0x0" ::: "memory");
#endif
#endif
}

// ---------------------------------------------------------------------------
// A-matrix fragment from a hoisted per-lane LDS pointer.
// p already includes (lane&15)*ldp + 16*(lane>>4); kofs is the K-tile offset.
// elems 0..15 -> K=16*hi+0..15 (matches pack_b B-fragment order).
// ---------------------------------------------------------------------------
__device__ __forceinline__ v16bf lds_a_frag(const bf16* p, int kofs) {
  v8bf lo = *(const v8bf*)(p + kofs);
  v8bf hv = *(const v8bf*)(p + kofs + 8);
  v16bf r;
#pragma unroll
  for (int e = 0; e < 8; ++e) { r[e] = lo[e]; r[e + 8] = hv[e]; }
  return r;
}

__device__ __forceinline__ float sig_f(float x) {
  return 1.0f / (1.0f + __expf(-x));
}
__device__ __forceinline__ float tanh_f(float x) {
  float e = __expf(-2.0f * x);
  return (1.0f - e) / (1.0f + e);
}

// ---------------------------------------------------------------------------
__global__ void init_state(bf16* __restrict__ h, float* __restrict__ c, int n) {
  int i = blockIdx.x * blockDim.x + threadIdx.x;
  if (i < n) { h[i] = (bf16)0.0f; c[i] = 0.0f; }
}

// ---------------------------------------------------------------------------
// Embedding gather -> time-major bf16, E padded to 320.
// ---------------------------------------------------------------------------
__global__ void embed_cast(const int* __restrict__ ids,
                           const float* __restrict__ emb,
                           bf16* __restrict__ X) {
  size_t idx = (size_t)blockIdx.x * blockDim.x + threadIdx.x;
  if (idx >= (size_t)TB * EPAD) return;
  int    e   = (int)(idx % EPAD);
  size_t row = idx / EPAD;          // row = t*BATCH + b
  int b = (int)(row % BATCH);
  int t = (int)(row / BATCH);
  float v = 0.0f;
  if (e < EMBED) v = emb[(size_t)ids[b * SEQ + t] * EMBED + e];
  X[idx] = (bf16)v;
}

// ---------------------------------------------------------------------------
// Pack row-major fp32 [K x N] into bf16 WMMA B-fragments:
// dst[nt][kt][lane][e] (32 B contiguous per lane); lane l -> col nt*16+(l&15),
// elem e -> K = kt*32 + 16*(l>>4) + e. K >= Kreal zero-padded.
// ---------------------------------------------------------------------------
__global__ void pack_b(const float* __restrict__ W, bf16* __restrict__ dst,
                       int Kreal, int Ktiles, int N) {
  int idx = blockIdx.x * blockDim.x + threadIdx.x;
  int total = (N / 16) * Ktiles * 32;
  if (idx >= total) return;
  int lane = idx & 31;
  int t    = idx >> 5;
  int kt   = t % Ktiles;
  int nt   = t / Ktiles;
  int n     = nt * 16 + (lane & 15);
  int kbase = kt * 32 + 16 * (lane >> 4);
  bf16* out = dst + (size_t)idx * 16;
#pragma unroll
  for (int e = 0; e < 16; ++e) {
    int k = kbase + e;
    float v = (k < Kreal) ? W[(size_t)k * N + n] : 0.0f;
    out[e] = (bf16)v;
  }
}

// ---------------------------------------------------------------------------
// GEMM1: XZp = X[TB x 320] * Wk + bias (bf16, fragment-packed layout).
// Wave computes 32(M) x 64(N); fully unrolled, ping-pong pipelined K loop with
// hoisted base pointers (fragment addresses are base + constant offsets).
// ---------------------------------------------------------------------------
__global__ void __launch_bounds__(256)
gemm1(const bf16* __restrict__ X, const bf16* __restrict__ Wp,
      const float* __restrict__ bias, bf16* __restrict__ XZp) {
  __shared__ __attribute__((aligned(32))) bf16 sA[32 * LDSP_G];

  const int lane = threadIdx.x & 31;
  const int wv   = threadIdx.x >> 5;
  const int mp   = blockIdx.x >> 2;                 // 4096 32-row tiles
  const int ng   = (blockIdx.x & 3) * 8 + wv;       // 32 groups of 64 N-cols
  const int KT   = EPAD / 32;                       // 10

  // Stage the 32 x 320 A tile (contiguous in X) into LDS, padded rows.
  {
    const bf16* Abase = X + (size_t)mp * 32 * EPAD;
    for (int c = threadIdx.x; c < 32 * (EPAD / 8); c += blockDim.x) {
      int row = c / (EPAD / 8);
      int col = (c % (EPAD / 8)) * 8;
      cp_b128(sA + row * LDSP_G + col, Abase + (size_t)row * EPAD + col);
    }
    cp_wait();
  }
  __syncthreads();

  const int nl = lane & 15;
  v8f acc[2][4];
#pragma unroll
  for (int j = 0; j < 4; ++j) {
    float bv = bias[ng * 64 + j * 16 + nl];
#pragma unroll
    for (int ml = 0; ml < 2; ++ml)
#pragma unroll
      for (int r = 0; r < 8; ++r) acc[ml][j][r] = bv;
  }

  // Hoisted per-lane base pointers; all loop offsets are compile-time consts.
  const bf16* pB = Wp + (size_t)ng * 4 * KT * 512 + lane * 16; // +(j*KT+kt)*512
  const bf16* pA = sA + (lane & 15) * LDSP_G + 16 * (lane >> 4); // +ml*16*LDSP_G+kt*32

  v16bf ab[2][2], bb[2][4];   // [parity][...]
#pragma unroll
  for (int ml = 0; ml < 2; ++ml) ab[0][ml] = lds_a_frag(pA + ml * 16 * LDSP_G, 0);
#pragma unroll
  for (int j = 0; j < 4; ++j) bb[0][j] = *(const v16bf*)(pB + (size_t)j * KT * 512);

#pragma unroll
  for (int kt = 0; kt < KT; ++kt) {
    const int cur = kt & 1, nxt = cur ^ 1;
    if (kt + 1 < KT) {
#pragma unroll
      for (int ml = 0; ml < 2; ++ml)
        ab[nxt][ml] = lds_a_frag(pA + ml * 16 * LDSP_G, (kt + 1) * 32);
#pragma unroll
      for (int j = 0; j < 4; ++j)
        bb[nxt][j] = *(const v16bf*)(pB + ((size_t)j * KT + kt + 1) * 512);
    }
#pragma unroll
    for (int ml = 0; ml < 2; ++ml)
#pragma unroll
      for (int j = 0; j < 4; ++j)
        acc[ml][j] = __builtin_amdgcn_wmma_f32_16x16x32_bf16(
            false, ab[cur][ml], false, bb[cur][j], (short)0, acc[ml][j], false, false);
  }

  // Fragment-packed store: one 16 B store per (ml, j), fully coalesced.
#pragma unroll
  for (int ml = 0; ml < 2; ++ml) {
    size_t gmt = (size_t)mp * 2 + ml;
#pragma unroll
    for (int j = 0; j < 4; ++j) {
      int nt = ng * 4 + j;
      v8bf pk;
#pragma unroll
      for (int r = 0; r < 8; ++r) pk[r] = (bf16)acc[ml][j][r];
      *(v8bf*)(XZp + ((gmt * 128 + nt) * 32 + lane) * 8) = pk;
    }
  }
}

// ---------------------------------------------------------------------------
// One LSTM timestep: async-stage the shared 16-row H tile to LDS, then each
// wave computes all four gate tiles for its (batch,unit) block with a fully
// unrolled ping-pong WMMA pipeline and applies the cell update in registers.
// ---------------------------------------------------------------------------
__global__ void __launch_bounds__(256)
lstm_step(const bf16* __restrict__ XZt, const bf16* __restrict__ Rp,
          const bf16* __restrict__ Hin, bf16* __restrict__ Hout,
          float* __restrict__ Cp, float* __restrict__ Hf32, int last) {
  __shared__ __attribute__((aligned(32))) bf16 sA[16 * LDSP_S];

  const int lane  = threadIdx.x & 31;
  const int wv    = threadIdx.x >> 5;
  const int mtile = blockIdx.x >> 2;                // 16 batch tiles
  const int ut    = (blockIdx.x & 3) * 8 + wv;      // 32 unit tiles
  const int KT    = UNITS / 32;                     // 16

  // Stage H tile (16 x 512, contiguous) into LDS.
  {
    const bf16* Abase = Hin + (size_t)mtile * 16 * UNITS;
    for (int c = threadIdx.x; c < 16 * (UNITS / 8); c += blockDim.x) {
      int row = c / (UNITS / 8);
      int col = (c % (UNITS / 8)) * 8;
      cp_b128(sA + row * LDSP_S + col, Abase + (size_t)row * UNITS + col);
    }
    cp_wait();
  }
  __syncthreads();

  // Accumulators preloaded with fragment-packed xz_t (bias already folded in).
  const bf16* pXZ = XZt + (((size_t)mtile * 128 + ut) * 32 + lane) * 8;
  v8f acc[4];
#pragma unroll
  for (int g = 0; g < 4; ++g) {
    v8bf xz = *(const v8bf*)(pXZ + (size_t)g * 32 * 32 * 8);
#pragma unroll
    for (int r = 0; r < 8; ++r) acc[g][r] = (float)xz[r];
  }

  // Hoisted base pointers; fragment addresses are base + constant offsets.
  const bf16* pB = Rp + (size_t)ut * KT * 512 + lane * 16; // +(g*32*KT+kt)*512
  const bf16* pA = sA + (lane & 15) * LDSP_S + 16 * (lane >> 4); // +kt*32

  v16bf ab[2], bb[2][4];
  ab[0] = lds_a_frag(pA, 0);
#pragma unroll
  for (int g = 0; g < 4; ++g)
    bb[0][g] = *(const v16bf*)(pB + (size_t)g * 32 * KT * 512);

#pragma unroll
  for (int kt = 0; kt < KT; ++kt) {
    const int cur = kt & 1, nxt = cur ^ 1;
    if (kt + 1 < KT) {
      ab[nxt] = lds_a_frag(pA, (kt + 1) * 32);
#pragma unroll
      for (int g = 0; g < 4; ++g)
        bb[nxt][g] = *(const v16bf*)(pB + ((size_t)g * 32 * KT + kt + 1) * 512);
    }
#pragma unroll
    for (int g = 0; g < 4; ++g)
      acc[g] = __builtin_amdgcn_wmma_f32_16x16x32_bf16(
          false, ab[cur], false, bb[cur][g], (short)0, acc[g], false, false);
  }

  // Gate epilogue: lane owns 8 (b,u) cells. C is fragment-packed (32 B/lane).
  const int hi = lane >> 4, nl = lane & 15;
  const int u  = ut * 16 + nl;
  size_t coff = (((size_t)mtile * 32 + ut) * 32 + lane) * 8;
  v8f cold = *(const v8f*)(Cp + coff);
  v8f cnew;
#pragma unroll
  for (int r = 0; r < 8; ++r) {
    int b = mtile * 16 + r + 8 * hi;
    float ig = sig_f(acc[0][r]);
    float fg = sig_f(acc[1][r]);
    float gg = tanh_f(acc[2][r]);
    float og = sig_f(acc[3][r]);
    float cn = fg * cold[r] + ig * gg;
    cnew[r] = cn;
    float hn = og * tanh_f(cn);
    Hout[(size_t)b * UNITS + u] = (bf16)hn;   // row-major for next step's A tile
    if (last) Hf32[(size_t)b * UNITS + u] = hn;
  }
  *(v8f*)(Cp + coff) = cnew;
}

// ---------------------------------------------------------------------------
__global__ void dense_softmax(const float* __restrict__ H,
                              const float* __restrict__ Wd,
                              const float* __restrict__ bd,
                              float* __restrict__ out) {
  int b = blockIdx.x * blockDim.x + threadIdx.x;
  if (b >= BATCH) return;
  float l0 = bd[0], l1 = bd[1];
  for (int u = 0; u < UNITS; ++u) {
    float h = H[(size_t)b * UNITS + u];
    l0 += h * Wd[u * 2 + 0];
    l1 += h * Wd[u * 2 + 1];
  }
  float m = fmaxf(l0, l1);
  float e0 = __expf(l0 - m), e1 = __expf(l1 - m);
  float s = e0 + e1;
  out[b * 2 + 0] = e0 / s;
  out[b * 2 + 1] = e1 / s;
}

// ---------------------------------------------------------------------------
extern "C" void kernel_launch(void* const* d_in, const int* in_sizes, int n_in,
                              void* d_out, int out_size, void* d_ws, size_t ws_size,
                              hipStream_t stream) {
  const int*   ids  = (const int*)  d_in[0];
  const float* emb  = (const float*)d_in[1];
  const float* Wk   = (const float*)d_in[2];
  const float* Rk   = (const float*)d_in[3];
  const float* bias = (const float*)d_in[4];
  const float* Wd   = (const float*)d_in[5];
  const float* bd   = (const float*)d_in[6];
  float* out = (float*)d_out;

  char* ws = (char*)d_ws;
  size_t off = 0;
  auto carve = [&](size_t bytes) -> void* {
    void* p = ws + off;
    off += (bytes + 255) & ~(size_t)255;
    return p;
  };

  bf16*  X     = (bf16*) carve((size_t)TB * EPAD * sizeof(bf16));             // 80 MB
  bf16*  Wkp   = (bf16*) carve((size_t)(NGATE/16) * (EPAD/32)  * 512 * 2);    // 1.25 MB
  bf16*  Rp    = (bf16*) carve((size_t)(NGATE/16) * (UNITS/32) * 512 * 2);    // 2 MB
  bf16*  XZp   = (bf16*) carve((size_t)TB * NGATE * sizeof(bf16));            // 512 MB
  bf16*  Hping = (bf16*) carve((size_t)BATCH * UNITS * sizeof(bf16));
  bf16*  Hpong = (bf16*) carve((size_t)BATCH * UNITS * sizeof(bf16));
  float* Cp    = (float*)carve((size_t)BATCH * UNITS * sizeof(float));
  float* Hf    = (float*)carve((size_t)BATCH * UNITS * sizeof(float));

  // h0 = c0 = 0 (every call; ws is not re-cleared between replays)
  init_state<<<(BATCH * UNITS + 255) / 256, 256, 0, stream>>>(Hping, Cp, BATCH * UNITS);

  {
    size_t tot = (size_t)TB * EPAD;
    embed_cast<<<(unsigned)((tot + 255) / 256), 256, 0, stream>>>(ids, emb, X);
  }

  pack_b<<<((NGATE/16) * (EPAD/32)  * 32 + 255) / 256, 256, 0, stream>>>(Wk, Wkp, EMBED, EPAD/32,  NGATE);
  pack_b<<<((NGATE/16) * (UNITS/32) * 32 + 255) / 256, 256, 0, stream>>>(Rk, Rp,  UNITS, UNITS/32, NGATE);

  // Input projection: 4096 32-row tiles x 4 blocks each = 16384 blocks.
  gemm1<<<16384, 256, 0, stream>>>(X, Wkp, bias, XZp);

  // Sequential scan: 512 stream-ordered step kernels, h ping-pong.
  for (int t = 0; t < SEQ; ++t) {
    const bf16* xzt = XZp + (size_t)t * BATCH * NGATE;
    const bf16* hin = (t & 1) ? Hpong : Hping;
    bf16*       hou = (t & 1) ? Hping : Hpong;
    lstm_step<<<64, 256, 0, stream>>>(xzt, Rp, hin, hou, Cp, Hf, (t == SEQ - 1) ? 1 : 0);
  }

  dense_softmax<<<1, 256, 0, stream>>>(Hf, Wd, bd, out);
}
